// GraphAttention_76261439308077
// MI455X (gfx1250) — matrix operand order repaired and checked by reading, not compiled, not run on hardware
//
#include <hip/hip_runtime.h>
#include <hip/hip_bf16.h>

// ---------------------------------------------------------------------------
// MI455X (gfx1250) fused GAT layer.
//  - All GEMMs use v_wmma_f32_16x16x32_bf16 (f32 accumulate, bf16 operands
//    staged in LDS) -- the throughput-optimal dense path on CDNA5.
//  - LDS layouts give ds_load_b128 for every WMMA fragment: A row-major
//    (two contiguous 16B runs), B TRANSPOSED [col][k] (two contiguous 16B runs).
//  - B staging processes two k-rows at once so f32->bf16 conversion uses
//    packed v_cvt_pk_bf16_f32 and 4B ds_store_b32 stores.
//  - Edge-hidden GEMMs fused with attention epilogues (no 614MB intermediate).
//  - Node tables Pi/Pj/Mj (~64MB) are L2-resident (192MB L2).
// ---------------------------------------------------------------------------

typedef __bf16 bf16_t;
typedef __bf16 v16bf __attribute__((ext_vector_type(16)));
typedef __bf16 v4bf  __attribute__((ext_vector_type(4)));
typedef __bf16 v2bf  __attribute__((ext_vector_type(2)));
typedef float  v8f   __attribute__((ext_vector_type(8)));

#define N_NODES 25000
#define N_EDGES 400000

__device__ __forceinline__ unsigned encF(float f) {
    unsigned u = __float_as_uint(f);
    return (u & 0x80000000u) ? ~u : (u | 0x80000000u);
}
__device__ __forceinline__ float decF(unsigned e) {
    unsigned u = (e & 0x80000000u) ? (e & 0x7FFFFFFFu) : ~e;
    return __uint_as_float(u);
}

// ---- WMMA fragment loaders (ISA 7.12.2 layouts, wave32) --------------------
template <int LD>
__device__ __forceinline__ v16bf load_a_frag(const bf16_t* As, int arow,
                                             int half, int kk) {
    union { uint4 q[2]; v16bf v; } u;
    const bf16_t* p = As + arow * LD + kk + half * 8;
    u.q[0] = *(const uint4*)(p);        // k = kk+half*8   .. +7
    u.q[1] = *(const uint4*)(p + 16);   // k = kk+16+half*8.. +7
    return u.v;
}
template <int LD>
__device__ __forceinline__ v16bf load_b_frag(const bf16_t* BsT, int col,
                                             int half, int kk) {
    union { uint4 q[2]; v16bf v; } u;
    const bf16_t* p = BsT + col * LD + kk + half * 8;
    u.q[0] = *(const uint4*)(p);        // k = kk+half*8   .. +7
    u.q[1] = *(const uint4*)(p + 16);   // k = kk+16+half*8.. +7
    return u.v;
}

__device__ __forceinline__ void store4(bf16_t* dst, float x0, float x1,
                                       float x2, float x3) {
    v4bf pk = {(bf16_t)x0, (bf16_t)x1, (bf16_t)x2, (bf16_t)x3};
    *(v4bf*)dst = pk;   // 2x v_cvt_pk_bf16_f32 + ds_store_b64
}
// Store a (k, k+1) pair for transposed-B staging: one packed cvt + b32 store.
__device__ __forceinline__ void store_pair(bf16_t* dst, float a, float b) {
    v2bf pk = {(bf16_t)a, (bf16_t)b};
    *(v2bf*)dst = pk;   // v_cvt_pk_bf16_f32 + ds_store_b32
}

// ---------------------------------------------------------------------------
// Generic GEMM: C[M,N] = (A ./ denom?) @ B + bias?.  Block tile 128x64,
// K-step 32, 8 waves (wave w owns rows w*16.., 4 col-tiles of 16).
// denom != nullptr => per-row/per-head (k>=64) normalization of A.
// ---------------------------------------------------------------------------
__global__ __launch_bounds__(256) void gemm_bf16_kernel(
    const float* __restrict__ A, const float* __restrict__ B,
    const float* __restrict__ bias, const float* __restrict__ denom,
    float* __restrict__ C, int M, int N, int K) {
    __shared__ __attribute__((aligned(16))) bf16_t As[128][40];   // rows x k
    __shared__ __attribute__((aligned(16))) bf16_t BsT[64][40];   // cols x k

    const int t = threadIdx.x;
    const int lane = t & 31;
    const int wave = t >> 5;
    const int half = lane >> 4;
    const int l15 = lane & 15;
    const int r0 = blockIdx.x * 128;
    const int n0 = blockIdx.y * 64;

    v8f acc[4];
#pragma unroll
    for (int j = 0; j < 4; ++j) acc[j] = {0.f, 0.f, 0.f, 0.f, 0.f, 0.f, 0.f, 0.f};

    for (int k0 = 0; k0 < K; k0 += 32) {
        __syncthreads();
#pragma unroll
        for (int i = 0; i < 4; ++i) {       // A tile 128x32, row-major in LDS
            int idx = t + 256 * i;
            int row = idx >> 3;
            int c4 = (idx & 7) << 2;
            int grow = r0 + row;
            float x0 = 0.f, x1 = 0.f, x2 = 0.f, x3 = 0.f;
            if (grow < M) {
                const float4 v = *(const float4*)(A + (size_t)grow * K + (k0 + c4));
                float sc = 1.0f;
                if (denom) {
                    float d = denom[2 * grow + ((k0 + c4) >> 6)];
                    sc = (d > 0.f) ? __builtin_amdgcn_rcpf(d) : 0.f;
                }
                x0 = v.x * sc; x1 = v.y * sc; x2 = v.z * sc; x3 = v.w * sc;
            }
            store4(&As[row][c4], x0, x1, x2, x3);
        }
        {                                   // B tile 32x64 -> transposed LDS
            int rp = t >> 4;                // k-row pair 0..15
            int c4 = (t & 15) << 2;         // col within tile
            const float* bp = B + (size_t)(k0 + 2 * rp) * N + (n0 + c4);
            const float4 v0 = *(const float4*)(bp);
            const float4 v1 = *(const float4*)(bp + N);
            store_pair(&BsT[c4 + 0][2 * rp], v0.x, v1.x);
            store_pair(&BsT[c4 + 1][2 * rp], v0.y, v1.y);
            store_pair(&BsT[c4 + 2][2 * rp], v0.z, v1.z);
            store_pair(&BsT[c4 + 3][2 * rp], v0.w, v1.w);
        }
        __syncthreads();

        v16bf af = load_a_frag<40>(&As[0][0], wave * 16 + l15, half, 0);
#pragma unroll
        for (int j = 0; j < 4; ++j) {
            v16bf bfr = load_b_frag<40>(&BsT[0][0], j * 16 + l15, half, 0);
            acc[j] = __builtin_amdgcn_wmma_f32_16x16x32_bf16(
                false, af, false, bfr, (short)0, acc[j], false, false);
        }
    }

#pragma unroll
    for (int j = 0; j < 4; ++j) {
        int col = n0 + j * 16 + l15;
        float bv = bias ? bias[col] : 0.f;
#pragma unroll
        for (int v = 0; v < 8; ++v) {
            int row = r0 + wave * 16 + half * 8 + v;
            if (row < M) C[(size_t)row * N + col] = acc[j][v] + bv;
        }
    }
}

// ---------------------------------------------------------------------------
// Edge pass 1: hidden = EF@e_W + Pi[dst] + Pj[src]; LeakyReLU; per-head
// logit = dot(hidden_h, attn_proj[h]); scatter-max per (dst, head).
// Block: 64 edges, 8 waves = 4 edge-tiles x 2 heads; K=64 (2 WMMA k-steps).
// ---------------------------------------------------------------------------
__global__ __launch_bounds__(256) void edge_logits_kernel(
    const float* __restrict__ EF, const int* __restrict__ dstI,
    const int* __restrict__ srcI, const float* __restrict__ Pi,
    const float* __restrict__ Pj, const float* __restrict__ attn,
    const float* __restrict__ eW, float* __restrict__ logits,
    unsigned* __restrict__ nodeMaxEnc) {
    __shared__ __attribute__((aligned(16))) bf16_t As[64][72];    // edges x K
    __shared__ __attribute__((aligned(16))) bf16_t BsT[256][72];  // cols x K

    const int t = threadIdx.x;
    const int lane = t & 31;
    const int wave = t >> 5;
    const int half = lane >> 4;
    const int l15 = lane & 15;
    const int mtile = wave >> 1;
    const int head = wave & 1;
    const int n_off = head * 128;
    const int e0 = blockIdx.x * 64;

#pragma unroll
    for (int i = 0; i < 4; ++i) {           // EF tile 64x64, row-major
        int idx = t + 256 * i;
        int row = idx >> 4;
        int c4 = (idx & 15) << 2;
        const float4 v = *(const float4*)(EF + (size_t)(e0 + row) * 64 + c4);
        store4(&As[row][c4], v.x, v.y, v.z, v.w);
    }
#pragma unroll
    for (int i = 0; i < 8; ++i) {           // e_W 64x256 -> transposed LDS
        int idx = t + 256 * i;              // 0..2047
        int rp = idx >> 6;                  // k-row pair 0..31
        int c4 = (idx & 63) << 2;           // col
        const float* bp = eW + (size_t)(2 * rp) * 256 + c4;
        const float4 v0 = *(const float4*)(bp);
        const float4 v1 = *(const float4*)(bp + 256);
        store_pair(&BsT[c4 + 0][2 * rp], v0.x, v1.x);
        store_pair(&BsT[c4 + 1][2 * rp], v0.y, v1.y);
        store_pair(&BsT[c4 + 2][2 * rp], v0.z, v1.z);
        store_pair(&BsT[c4 + 3][2 * rp], v0.w, v1.w);
    }
    __syncthreads();

    v8f acc[8];
#pragma unroll
    for (int j = 0; j < 8; ++j) acc[j] = {0.f, 0.f, 0.f, 0.f, 0.f, 0.f, 0.f, 0.f};

#pragma unroll
    for (int kk = 0; kk < 64; kk += 32) {
        v16bf af = load_a_frag<72>(&As[0][0], mtile * 16 + l15, half, kk);
#pragma unroll
        for (int j = 0; j < 8; ++j) {
            v16bf bfr = load_b_frag<72>(&BsT[0][0], n_off + j * 16 + l15, half, kk);
            acc[j] = __builtin_amdgcn_wmma_f32_16x16x32_bf16(
                false, af, false, bfr, (short)0, acc[j], false, false);
        }
    }

    const int ebase = e0 + mtile * 16 + half * 8;
    int dstv[8], srcv[8];
#pragma unroll
    for (int v = 0; v < 8; ++v) {
        dstv[v] = dstI[ebase + v];
        srcv[v] = srcI[ebase + v];
    }
    float partial[8] = {0.f, 0.f, 0.f, 0.f, 0.f, 0.f, 0.f, 0.f};
#pragma unroll
    for (int j = 0; j < 8; ++j) {
        int hc = n_off + j * 16 + l15;
        float pr = attn[hc];
#pragma unroll
        for (int v = 0; v < 8; ++v) {
            float h = acc[j][v] + Pi[dstv[v] * 256 + hc] + Pj[srcv[v] * 256 + hc];
            h = (h >= 0.f) ? h : 0.2f * h;
            partial[v] += h * pr;
        }
    }
#pragma unroll
    for (int m = 1; m < 16; m <<= 1)
#pragma unroll
        for (int v = 0; v < 8; ++v) partial[v] += __shfl_xor(partial[v], m, 32);
    if (l15 == 0) {
#pragma unroll
        for (int v = 0; v < 8; ++v) {
            logits[(ebase + v) * 2 + head] = partial[v];
            atomicMax(&nodeMaxEnc[dstv[v] * 2 + head], encF(partial[v]));
        }
    }
}

// ---------------------------------------------------------------------------
// Edge pass 2: msg = EF@msg_W_bot + Mj[src]; w = exp(logit - max[dst]);
// denom[dst,h] += w (once per edge); agg[dst,c] += w_h(c) * msg_c
// (softmax normalization deferred to the final GEMM).
// Block: 128 edges, one 16-edge tile per wave; K=64.
// ---------------------------------------------------------------------------
__global__ __launch_bounds__(256) void edge_msg_kernel(
    const float* __restrict__ EF, const int* __restrict__ dstI,
    const int* __restrict__ srcI, const float* __restrict__ Mj,
    const float* __restrict__ Wm, const float* __restrict__ logits,
    const unsigned* __restrict__ nodeMaxEnc, float* __restrict__ denom,
    float* __restrict__ agg) {
    __shared__ __attribute__((aligned(16))) bf16_t As[128][72];   // edges x K
    __shared__ __attribute__((aligned(16))) bf16_t BsT[128][72];  // cols x K

    const int t = threadIdx.x;
    const int lane = t & 31;
    const int wave = t >> 5;
    const int half = lane >> 4;
    const int l15 = lane & 15;
    const int e0 = blockIdx.x * 128;

#pragma unroll
    for (int i = 0; i < 8; ++i) {           // EF tile 128x64, row-major
        int idx = t + 256 * i;
        int row = idx >> 4;
        int c4 = (idx & 15) << 2;
        const float4 v = *(const float4*)(EF + (size_t)(e0 + row) * 64 + c4);
        store4(&As[row][c4], v.x, v.y, v.z, v.w);
    }
#pragma unroll
    for (int i = 0; i < 4; ++i) {           // Wm 64x128 -> transposed LDS
        int idx = t + 256 * i;              // 0..1023
        int rp = idx >> 5;                  // k-row pair 0..31
        int c4 = (idx & 31) << 2;           // col
        const float* bp = Wm + (size_t)(2 * rp) * 128 + c4;
        const float4 v0 = *(const float4*)(bp);
        const float4 v1 = *(const float4*)(bp + 128);
        store_pair(&BsT[c4 + 0][2 * rp], v0.x, v1.x);
        store_pair(&BsT[c4 + 1][2 * rp], v0.y, v1.y);
        store_pair(&BsT[c4 + 2][2 * rp], v0.z, v1.z);
        store_pair(&BsT[c4 + 3][2 * rp], v0.w, v1.w);
    }
    __syncthreads();

    v8f acc[8];
#pragma unroll
    for (int j = 0; j < 8; ++j) acc[j] = {0.f, 0.f, 0.f, 0.f, 0.f, 0.f, 0.f, 0.f};

#pragma unroll
    for (int kk = 0; kk < 64; kk += 32) {
        v16bf af = load_a_frag<72>(&As[0][0], wave * 16 + l15, half, kk);
#pragma unroll
        for (int j = 0; j < 8; ++j) {
            v16bf bfr = load_b_frag<72>(&BsT[0][0], j * 16 + l15, half, kk);
            acc[j] = __builtin_amdgcn_wmma_f32_16x16x32_bf16(
                false, af, false, bfr, (short)0, acc[j], false, false);
        }
    }

    const int ebase = e0 + wave * 16 + half * 8;
    int dn[8], sv[8];
    float w0[8], w1[8];
#pragma unroll
    for (int v = 0; v < 8; ++v) {
        int e = ebase + v;
        int d = dstI[e];
        dn[v] = d;
        sv[v] = srcI[e];
        w0[v] = __expf(logits[e * 2 + 0] - decF(nodeMaxEnc[d * 2 + 0]));
        w1[v] = __expf(logits[e * 2 + 1] - decF(nodeMaxEnc[d * 2 + 1]));
    }
    if (l15 == 0) {
#pragma unroll
        for (int v = 0; v < 8; ++v) {
            atomicAdd(&denom[dn[v] * 2 + 0], w0[v]);
            atomicAdd(&denom[dn[v] * 2 + 1], w1[v]);
        }
    }
#pragma unroll
    for (int j = 0; j < 8; ++j) {
        int c = j * 16 + l15;               // j<4 -> head0, j>=4 -> head1
#pragma unroll
        for (int v = 0; v < 8; ++v) {
            float val = acc[j][v] + Mj[sv[v] * 128 + c];
            float w = (j < 4) ? w0[v] : w1[v];
            atomicAdd(&agg[dn[v] * 128 + c], val * w);
        }
    }
}

// ---------------------------------------------------------------------------
__global__ void bias_sum_kernel(const float* a, const float* b, const float* c,
                                float* o, int n) {
    int i = blockIdx.x * blockDim.x + threadIdx.x;
    if (i < n) o[i] = a[i] + b[i] + c[i];
}

__global__ void init_kernel(float* agg, float* denom, unsigned* maxEnc,
                            int nAgg, int nND) {
    int i = blockIdx.x * blockDim.x + threadIdx.x;
    if (i < nAgg) agg[i] = 0.f;
    if (i < nND) { denom[i] = 0.f; maxEnc[i] = 0u; }
}

// ---------------------------------------------------------------------------
extern "C" void kernel_launch(void* const* d_in, const int* in_sizes, int n_in,
                              void* d_out, int out_size, void* d_ws,
                              size_t ws_size, hipStream_t stream) {
    const float* NF    = (const float*)d_in[0];
    const float* EF    = (const float*)d_in[1];
    const int*   EIx   = (const int*)d_in[2];
    const float* ni_W  = (const float*)d_in[3];
    const float* ni_b  = (const float*)d_in[4];
    const float* nj_W  = (const float*)d_in[5];
    const float* nj_b  = (const float*)d_in[6];
    const float* e_W   = (const float*)d_in[7];
    const float* e_b   = (const float*)d_in[8];
    const float* attn  = (const float*)d_in[9];
    const float* msg_W = (const float*)d_in[10];
    const float* msg_b = (const float*)d_in[11];
    const float* out_W = (const float*)d_in[12];
    const float* out_b = (const float*)d_in[13];
    float* out = (float*)d_out;

    const int N = N_NODES, E = N_EDGES;
    float* ws = (float*)d_ws;
    float*    Pi     = ws;                                  // N*256
    float*    Pj     = Pi + (size_t)N * 256;                // N*256
    float*    Mj     = Pj + (size_t)N * 256;                // N*128
    float*    bias_h = Mj + (size_t)N * 128;                // 256
    float*    logits = bias_h + 256;                        // E*2
    unsigned* maxEnc = (unsigned*)(logits + (size_t)E * 2); // N*2
    float*    denom  = (float*)(maxEnc + (size_t)N * 2);    // N*2
    float*    agg    = denom + (size_t)N * 2;               // N*128

    const int* dstI = EIx;       // edge_index[0] = aggregation index
    const int* srcI = EIx + E;   // edge_index[1]

    bias_sum_kernel<<<1, 256, 0, stream>>>(ni_b, nj_b, e_b, bias_h, 256);
    init_kernel<<<(N * 128 + 255) / 256, 256, 0, stream>>>(agg, denom, maxEnc,
                                                           N * 128, N * 2);

    dim3 blk(256);
    int mg = (N + 127) / 128;
    // Node projections (hidden bias sum folded into Pi; msg_b into Mj).
    gemm_bf16_kernel<<<dim3(mg, 4), blk, 0, stream>>>(NF, ni_W, bias_h, nullptr,
                                                      Pi, N, 256, 128);
    gemm_bf16_kernel<<<dim3(mg, 4), blk, 0, stream>>>(NF, nj_W, nullptr, nullptr,
                                                      Pj, N, 256, 128);
    gemm_bf16_kernel<<<dim3(mg, 2), blk, 0, stream>>>(NF, msg_W, msg_b, nullptr,
                                                      Mj, N, 128, 128);
    // Fused edge passes.
    edge_logits_kernel<<<E / 64, blk, 0, stream>>>(EF, dstI, srcI, Pi, Pj, attn,
                                                   e_W, logits, maxEnc);
    edge_msg_kernel<<<E / 128, blk, 0, stream>>>(EF, dstI, srcI, Mj,
                                                 msg_W + 128 * 128, logits,
                                                 maxEnc, denom, agg);
    // Output GEMM with fused softmax normalization (denom, 0-guarded).
    gemm_bf16_kernel<<<dim3(mg, 2), blk, 0, stream>>>(agg, out_W, out_b, denom,
                                                      out, N, 128, 128);
}
